// MaxSimRanker_14010183320058
// MI455X (gfx1250) — compile-verified, matrix operand order checked by reading, not run on hardware
//
#include <hip/hip_runtime.h>

// ---------------- problem constants (match reference) ----------------
#define BS     16
#define QLEN   32
#define H      128
#define NDOCS  16384
#define DLEN   128
#define NPIDS  500
#define TOPK   100

#define QP     136   // Q LDS pitch in halves   (128 + 8 pad)
#define DPF    132   // D LDS pitch in floats   (128 + 4 pad)
#define DPB    5     // docs per block (Q reuse)

typedef __attribute__((ext_vector_type(16))) _Float16 v16h;
typedef __attribute__((ext_vector_type(8)))  float    v8f;
typedef int v4i __attribute__((vector_size(4 * sizeof(int))));

#if defined(__has_builtin)
#if __has_builtin(__builtin_amdgcn_global_load_async_to_lds_b128)
#define HAS_ASYNC_LDS 1
#endif
#endif
#ifndef HAS_ASYNC_LDS
#define HAS_ASYNC_LDS 0
#endif

// ---- async global->LDS b128 (per-lane 16B), ASYNCcnt tracked -----------
__device__ __forceinline__ void lds_copy_b128(const float* __restrict__ g,
                                              float* l) {
#if HAS_ASYNC_LDS
  __builtin_amdgcn_global_load_async_to_lds_b128(
      (v4i*)(uintptr_t)g, (v4i*)(uintptr_t)l, 0, 0);
#else
  float4 f = *(const float4*)g;
  *(float4*)l = f;
#endif
}

__device__ __forceinline__ void wait_async_le16() {
#if HAS_ASYNC_LDS
#if __has_builtin(__builtin_amdgcn_s_wait_asynccnt)
  __builtin_amdgcn_s_wait_asynccnt(16);
#else
  asm volatile("s_wait_asynccnt 0x10" ::: "memory");
#endif
  asm volatile("" ::: "memory");
#endif
}

__device__ __forceinline__ void wait_async_le0() {
#if HAS_ASYNC_LDS
#if __has_builtin(__builtin_amdgcn_s_wait_asynccnt)
  __builtin_amdgcn_s_wait_asynccnt(0);
#else
  asm volatile("s_wait_asynccnt 0x0" ::: "memory");
#endif
  asm volatile("" ::: "memory");
#endif
}

// ---------------------------------------------------------------------
// Kernel 1: range-mask + first-occurrence dedup of pids (per batch row)
// ---------------------------------------------------------------------
__global__ __launch_bounds__(512) void maxsim_dedup_kernel(
    const int* __restrict__ pids, int* __restrict__ pids_out) {
  __shared__ int row[NPIDS];
  const int b = blockIdx.x;
  const int t = threadIdx.x;
  if (t < NPIDS) {
    int p = pids[b * NPIDS + t];
    if (p < 0 || p >= NDOCS) p = -1;
    row[t] = p;
  }
  __syncthreads();
  if (t < NPIDS) {
    int p = row[t];
    if (p >= 0) {
      for (int j = 0; j < t; ++j) {
        if (row[j] == p) { p = -1; break; }
      }
    }
    pids_out[b * NPIDS + t] = p;
  }
}

// ---------------------------------------------------------------------
// Kernel 2: MaxSim scoring. One block = (batch b, DPB docs).
// 256 threads = 8 wave32. Wave w owns token rows/columns [16w, 16w+16):
// it async-stages exactly the D rows it will read -> only per-wave
// s_wait_asynccnt is needed for the double-buffered D tiles.
// Pipeline: HBM->L2 (prefetch dd+2), L2->LDS (async dd+1), LDS->WMMA (dd).
// ---------------------------------------------------------------------
__global__ __launch_bounds__(256) void maxsim_score_kernel(
    const float* __restrict__ qv,     // [BS][QLEN][H]
    const float* __restrict__ vecs,   // [NDOCS][DLEN][H]
    const int*   __restrict__ pidsD,  // [BS][NPIDS] deduped
    float*       __restrict__ scores) // [BS][NPIDS]
{
  __shared__ _Float16 Qs[QLEN * QP];       //  8.7 KB, f16
  __shared__ float    Dbuf[2][DLEN * DPF]; // 132 KB, fp32 double buffer
  __shared__ float    partialMax[8 * QLEN];
  __shared__ int      qnz[QLEN];

  const int b    = blockIdx.y;
  const int d0   = blockIdx.x * DPB;
  const int tid  = threadIdx.x;
  const int lane = tid & 31;
  const int w    = tid >> 5;

  // ---- stage Q (fp32 -> f16) once per block; build qmask flags ----
  if (tid < QLEN) qnz[tid] = 0;
  __syncthreads();
  {
    const int r = tid >> 3;            // 0..31 query row
    const int c = (tid & 7) * 16;      // 16 floats per thread
    const float* src = qv + ((size_t)b * QLEN + r) * H + c;
    unsigned nzb = 0;
#pragma unroll
    for (int i = 0; i < 16; i += 4) {
      float4 f = *(const float4*)(src + i);
      nzb |= (__float_as_uint(f.x) | __float_as_uint(f.y) |
              __float_as_uint(f.z) | __float_as_uint(f.w)) & 0x7fffffffu;
      Qs[r * QP + c + i + 0] = (_Float16)f.x;
      Qs[r * QP + c + i + 1] = (_Float16)f.y;
      Qs[r * QP + c + i + 2] = (_Float16)f.z;
      Qs[r * QP + c + i + 3] = (_Float16)f.w;
    }
    if (nzb) atomicOr(&qnz[r], 1);
  }
  __syncthreads();

  // ---- prologue: async-stage doc d0 into buffer 0 (wave-private rows) ----
  {
    const int pid0 = pidsD[b * NPIDS + d0];
    const int ps   = (pid0 < 0) ? 0 : pid0;   // dummy-stage invalid pids
    const float* g0 = vecs + ((size_t)ps * DLEN + (w << 4)) * H + (lane << 2);
    float*       l0 = &Dbuf[0][(w << 4) * DPF + (lane << 2)];
#pragma unroll
    for (int i = 0; i < 16; ++i)
      lds_copy_b128(g0 + (size_t)i * H, l0 + i * DPF);
  }

  for (int dd = 0; dd < DPB; ++dd) {
    const int di  = d0 + dd;
    const int pid = pidsD[b * NPIDS + di];     // block-uniform

    // ---- issue async stage of the NEXT doc into the other buffer ----
    if (dd + 1 < DPB) {
      const int pn = pidsD[b * NPIDS + di + 1];
      const int ps = (pn < 0) ? 0 : pn;
      const float* g0 = vecs + ((size_t)ps * DLEN + (w << 4)) * H + (lane << 2);
      float*       l0 = &Dbuf[(dd + 1) & 1][(w << 4) * DPF + (lane << 2)];
#pragma unroll
      for (int i = 0; i < 16; ++i)
        lds_copy_b128(g0 + (size_t)i * H, l0 + i * DPF);
      // ---- warm L2 for doc dd+2 (cacheline-granular prefetch) ----
      if (dd + 2 < DPB) {
        const int p2 = pidsD[b * NPIDS + di + 2];
        if (p2 >= 0) {
          const float* pbase =
              vecs + ((size_t)p2 * DLEN + (w << 4) + (lane & 15)) * H;
#pragma unroll
          for (int j = 0; j < 2; ++j)
            __builtin_prefetch(pbase + ((lane >> 4) + 2 * j) * 32, 0, 1);
        }
      }
      wait_async_le16();                       // current buffer resident
    } else {
      wait_async_le0();
    }

    if (pid < 0) {                             // masked pid -> -inf score
      if (tid == 0) scores[b * NPIDS + di] = -__builtin_inff();
      continue;                                // uniform branch: barrier-safe
    }

    const float* __restrict__ bufc = &Dbuf[dd & 1][0];

    // ---- WMMA: S[32 x 128] = Q[32 x 128] * D^T, K chained over 128 ----
    v8f c0 = {};                  // q rows 0..15, tokens 16w..16w+15
    v8f c1 = {};                  // q rows 16..31
    unsigned nzb = 0;             // OR of |bits| over this lane's K-half
    const int nrow = lane & 15;
    const int g    = lane >> 4;
    const int tok  = (w << 4) + nrow;          // this lane's token column
    const float* __restrict__ Drow = bufc + tok * DPF;
#pragma unroll
    for (int kb = 0; kb < 4; ++kb) {
      union { v16h v; _Float16 h[16]; unsigned u[8]; } A0, A1, B;
      const int base = kb * 32;
#pragma unroll
      for (int p = 0; p < 8; ++p) {
        // 16-bit A/B fragment layout: lanes 0-15 = K[0..7]+K[16..23],
        // lanes 16-31 = K[8..15]+K[24..31]; consecutive-K pairs per slot.
        const int off = base + ((p < 4) ? (8 * g + 2 * p)
                                        : (16 + 8 * g + 2 * (p - 4)));
        float2 f = *(const float2*)(Drow + off);
        nzb |= (__float_as_uint(f.x) | __float_as_uint(f.y)) & 0x7fffffffu;
        B.h[2 * p + 0] = (_Float16)f.x;
        B.h[2 * p + 1] = (_Float16)f.y;
        A0.u[p] = *(const unsigned*)&Qs[nrow * QP + off];
        A1.u[p] = *(const unsigned*)&Qs[(16 + nrow) * QP + off];
      }
      c0 = __builtin_amdgcn_wmma_f32_16x16x32_f16(false, A0.v, false, B.v,
                                                  (short)0, c0, false, false);
      c1 = __builtin_amdgcn_wmma_f32_16x16x32_f16(false, A1.v, false, B.v,
                                                  (short)0, c1, false, false);
    }

    // ---- dmask: lanes L and L+16 saw complementary K-halves of token tok
    nzb |= (unsigned)__shfl_xor((int)nzb, 16, 32);
    const bool dz = (nzb == 0);
    float m0[8], m1[8];
#pragma unroll
    for (int r = 0; r < 8; ++r) {
      m0[r] = dz ? -__builtin_inff() : c0[r];
      m1[r] = dz ? -__builtin_inff() : c1[r];
    }
    // max across the 16 lanes of each half (columns of this token tile)
#pragma unroll
    for (int m = 1; m <= 8; m <<= 1) {
#pragma unroll
      for (int r = 0; r < 8; ++r) {
        m0[r] = fmaxf(m0[r], __shfl_xor(m0[r], m, 32));
        m1[r] = fmaxf(m1[r], __shfl_xor(m1[r], m, 32));
      }
    }
    if ((lane & 15) == 0) {
      const int mb = g * 8;  // C layout: half 0 -> M=r, half 1 -> M=r+8
#pragma unroll
      for (int r = 0; r < 8; ++r) {
        partialMax[w * QLEN + mb + r]      = m0[r];
        partialMax[w * QLEN + 16 + mb + r] = m1[r];
      }
    }
    __syncthreads();

    // ---- combine 8 token tiles, apply qmask, sum over 32 query rows ----
    if (tid < QLEN) {
      float v = partialMax[tid];
#pragma unroll
      for (int ww = 1; ww < 8; ++ww)
        v = fmaxf(v, partialMax[ww * QLEN + tid]);
      if (qnz[tid] == 0) v = 0.f;
#pragma unroll
      for (int m = 16; m >= 1; m >>= 1) v += __shfl_xor(v, m, 32);
      if (tid == 0) scores[b * NPIDS + di] = v;
    }
    __syncthreads();   // partialMax reuse next iteration
  }
}

// ---------------------------------------------------------------------
// Kernel 3: per-batch sorted top-100 (bitonic, tie-break = lower index)
// ---------------------------------------------------------------------
__global__ __launch_bounds__(512) void maxsim_topk_kernel(
    const float* __restrict__ scores, const int* __restrict__ pidsD,
    float* __restrict__ out) {
  __shared__ float skey[512];
  __shared__ int   sidx[512];
  const int b = blockIdx.x;
  const int t = threadIdx.x;

  skey[t] = (t < NPIDS) ? scores[b * NPIDS + t] : -__builtin_inff();
  sidx[t] = t;

  for (int k2 = 2; k2 <= 512; k2 <<= 1) {
    for (int j = k2 >> 1; j > 0; j >>= 1) {
      __syncthreads();
      const int ixj = t ^ j;
      if (ixj > t) {
        const float ka = skey[t], kb = skey[ixj];
        const int   ia = sidx[t], ib = sidx[ixj];
        // "a ranks first" in final descending order (ties: lower index)
        const bool f    = (ka > kb) || (ka == kb && ia < ib);
        const bool desc = ((t & k2) == 0);
        if (desc ? !f : f) {
          skey[t] = kb; skey[ixj] = ka;
          sidx[t] = ib; sidx[ixj] = ia;
        }
      }
    }
  }
  __syncthreads();

  if (t < TOPK) {
    out[b * TOPK + t] = skey[t];
    const int idx = sidx[t];
    const int pid = (idx < NPIDS) ? pidsD[b * NPIDS + idx] : -1;
    out[BS * TOPK + b * TOPK + t] = (float)pid;   // BOUNDARY_LO == 0
  }
}

// ---------------------------------------------------------------------
extern "C" void kernel_launch(void* const* d_in, const int* in_sizes, int n_in,
                              void* d_out, int out_size, void* d_ws, size_t ws_size,
                              hipStream_t stream) {
  (void)in_sizes; (void)n_in; (void)out_size; (void)ws_size;
  const float* qv   = (const float*)d_in[0];   // [16][32][128] f32
  const float* vecs = (const float*)d_in[1];   // [16384][128][128] f32
  const int*   pids = (const int*)d_in[2];     // [16][500]
  // d_in[3] = k (always 100 in this benchmark; compile-time constant)

  int*   pidsD  = (int*)d_ws;                                       // 32000 B
  float* scores = (float*)((char*)d_ws + BS * NPIDS * sizeof(int)); // 32000 B

  maxsim_dedup_kernel<<<BS, 512, 0, stream>>>(pids, pidsD);

  dim3 grid(NPIDS / DPB, BS);
  maxsim_score_kernel<<<grid, 256, 0, stream>>>(qv, vecs, pidsD, scores);

  maxsim_topk_kernel<<<BS, 512, 0, stream>>>(scores, pidsD, (float*)d_out);
}
